// WaveNet_240518168893
// MI455X (gfx1250) — matrix operand order repaired and checked by reading, not compiled, not run on hardware
//
#include <hip/hip_runtime.h>
#include <hip/hip_bf16.h>

// ---------------------------------------------------------------------------
// WaveNet on MI455X (gfx1250): all 16 layers fused per time-tile, activations
// resident in LDS, channel matmuls on V_WMMA_F32_16X16X32_F16 (wave32).
// 512 threads = 16 waves/WGP keeps every wave under 256 VGPRs (no MSB mode).
// 1-channel projections use packed-f16 VALU (v_pk_fma_f16).
// ---------------------------------------------------------------------------

typedef __attribute__((ext_vector_type(16))) _Float16 v16h;
typedef __attribute__((ext_vector_type(8)))  _Float16 v8h;
typedef __attribute__((ext_vector_type(8)))  float    v8f;

#define WN_B     8
#define WN_T     65536
#define RES_CH   32
#define SKIP_CH  64
#define KS       3
#define NLAYER   16

#define TILE     512          // output timesteps per workgroup
#define HALO     1024         // receptive field = 1020, rounded up
#define EXT      (TILE + HALO)            // 1536
#define NTHREADS 512
#define NWAVES   16
#define NT_PER_WAVE (EXT / 16 / NWAVES)   // 6 WMMA N-tiles per wave
#define SKIP_NT     (TILE / 16 / NWAVES)  // 2 skip N-tiles per wave

// packed-weight regions in d_ws (element counts, f16)
#define FG_TILE  512                              // 32 lanes * 16 halfs
#define N_FG     (NLAYER*2*2*KS*FG_TILE)          // 98304
#define OFF_WS   (N_FG)
#define N_WS     (NLAYER*4*FG_TILE)               // 32768
#define OFF_SK1  (OFF_WS + N_WS)                  // 131072
#define N_SK1    (4*2*FG_TILE)                    // 4096
#define N_PACK   (OFF_SK1 + N_SK1)                // 135168 halfs = 264 KiB
// after the f16 region: SKIP_CH floats = Σ_layers b_s  (bias folded once)
#define N_TOTAL  (N_PACK + 2*SKIP_CH)             // bias sum as 64 floats

#define SMEM_BYTES (2*EXT*RES_CH*2 + EXT*4)       // 198 KiB dynamic LDS

// A-operand K index for (lane, j), contiguous K order matching B:
// lanes 0-15 hold K {0..15}; lanes 16-31 hold K {16..31}
__host__ __device__ inline int kmap(int lane, int j) {
  return j + ((lane >> 4) ? 16 : 0);
}

// ---------------------------------------------------------------------------
// Weight packing: f32 -> f16 in exact WMMA A-operand lane layout, plus the
// layer-summed skip bias (linear => fold once at the end).
// ---------------------------------------------------------------------------
__global__ void wn_pack(const float* __restrict__ Wf, const float* __restrict__ Wg,
                        const float* __restrict__ Ws, const float* __restrict__ Wsk1,
                        const float* __restrict__ bs, _Float16* __restrict__ wp)
{
  int idx = blockIdx.x * blockDim.x + threadIdx.x;
  if (idx >= N_PACK) {
    int c = idx - N_PACK;
    if (c < SKIP_CH) {
      float s = 0.f;
      for (int l = 0; l < NLAYER; ++l) s += bs[l*SKIP_CH + c];
      ((float*)(wp + N_PACK))[c] = s;
    }
    return;
  }
  int j    = idx & 15;
  int lane = (idx >> 4) & 31;
  int tile = idx >> 9;
  int ci   = kmap(lane, j);
  int mrow = lane & 15;
  float v;
  if (idx < OFF_WS) {                 // W_f / W_g : tiles (layer, gate, mt, k)
    int k     = tile % KS;
    int mt    = (tile / KS) & 1;
    int gate  = (tile / (KS*2)) & 1;
    int layer = tile / (KS*4);
    int m = mt*16 + mrow;
    const float* W = gate ? Wg : Wf;
    v = W[((layer*RES_CH + m)*RES_CH + ci)*KS + k];
  } else if (idx < OFF_SK1) {         // W_s : tiles (layer, mt)
    int rt = tile - (OFF_WS >> 9);
    int mt = rt & 3;
    int layer = rt >> 2;
    int m = mt*16 + mrow;
    v = Ws[(layer*SKIP_CH + m)*RES_CH + ci];
  } else {                            // W_sk1 : tiles (mt, kc)
    int rt = tile - (OFF_SK1 >> 9);
    int kc = rt & 1;
    int mt = rt >> 1;
    int m = mt*16 + mrow;
    v = Wsk1[m*SKIP_CH + kc*32 + ci];
  }
  wp[idx] = (_Float16)v;
}

__device__ inline float fast_sigmoid(float x) { return 1.f / (1.f + __expf(-x)); }
__device__ inline float fast_tanh(float x) {
  float e = __expf(-2.f * x);
  return (1.f - e) / (1.f + e);
}

__device__ inline v8f wmma_f16(v16h a, v16h b, v8f c) {
  return __builtin_amdgcn_wmma_f32_16x16x32_f16(false, a, false, b, (short)0, c,
                                                false, false);
}

// horizontal sum of a packed-f16 octet -> f32 (one convert at the end)
__device__ inline float hsum8h(v8h p) {
  _Float16 a = (_Float16)(p[0] + p[4]) + (_Float16)(p[1] + p[5]);
  _Float16 b = (_Float16)(p[2] + p[6]) + (_Float16)(p[3] + p[7]);
  return (float)(a + b);
}

// ---------------------------------------------------------------------------
// Fused 16-layer WaveNet tile kernel.
// ---------------------------------------------------------------------------
__global__ void __launch_bounds__(NTHREADS)
wn_fused(const float* __restrict__ x,
         const float* __restrict__ W_in, const float* __restrict__ b_in,
         const float* __restrict__ b_f,  const float* __restrict__ b_g,
         const float* __restrict__ W_o,  const float* __restrict__ b_o,
         const float* __restrict__ b_sk1,
         const float* __restrict__ W_sk2,const float* __restrict__ b_sk2,
         const _Float16* __restrict__ wp,
         float* __restrict__ outp)
{
  extern __shared__ char smem[];
  _Float16* resid = (_Float16*)smem;                        // [EXT][32] f16
  _Float16* gated = (_Float16*)(smem + EXT*RES_CH*2);       // [EXT][32] f16
  float*    outb  = (float*)(smem + 2*EXT*RES_CH*2);        // [EXT]     f32

  const int tid  = threadIdx.x;
  const int wave = tid >> 5;
  const int lane = tid & 31;
  const int hi   = lane >> 4;        // K half: lanes 0-15 -> ch 0-15, 16-31 -> ch 16-31
  const int ln   = lane & 15;        // N (time) within WMMA tile

  const int tilesPerB = WN_T / TILE;               // 128
  const int bidx = blockIdx.x / tilesPerB;
  const int tile = blockIdx.x % tilesPerB;
  const long t0  = (long)tile * TILE;

  // --- load 1-channel input window (halo zero-padded) -----------------------
  for (int t = tid; t < EXT; t += NTHREADS) {
    long g = t0 - HALO + t;
    outb[t] = (g >= 0) ? x[(long)bidx*WN_T + g] : 0.f;
  }
  __syncthreads();

  // --- per-wave skip accumulator (64ch x 32t, f32, 64 VGPRs) ----------------
  v8f skipacc[SKIP_NT][4];
  #pragma unroll
  for (int nt = 0; nt < SKIP_NT; ++nt)
    #pragma unroll
    for (int mt = 0; mt < 4; ++mt)
      #pragma unroll
      for (int r = 0; r < 8; ++r) skipacc[nt][mt][r] = 0.f;

  const int c8 = (tid & 3) * 8;      // fixed 8-channel slice for Phase A

  for (int layer = 0; layer < NLAYER; ++layer) {
    const int d = 1 << (layer & 7);

    // --- Phase A: residual = W_in * out + b_in  (rank-1, VALU) --------------
    {
      float wi8[8], bi8[8];
      #pragma unroll
      for (int r = 0; r < 8; ++r) {
        wi8[r] = W_in[layer*RES_CH + c8 + r];
        bi8[r] = b_in[layer*RES_CH + c8 + r];
      }
      // idx & 3 == tid & 3 is loop-invariant (stride 512 ≡ 0 mod 4)
      #pragma unroll
      for (int i2 = 0; i2 < EXT*4/NTHREADS; ++i2) {
        int t = (tid + i2*NTHREADS) >> 2;
        float o = outb[t];
        v8h hv;
        #pragma unroll
        for (int r = 0; r < 8; ++r) hv[r] = (_Float16)(wi8[r]*o + bi8[r]);
        *(v8h*)(resid + t*RES_CH + c8) = hv;
      }
    }
    __syncthreads();

    // --- Phase B: f/g dilated convs via WMMA, gated = tanh(f)*sigmoid(g) ----
    // mt loop outermost: only 6 weight tiles (48 VGPRs) live at a time.
    {
      #pragma unroll
      for (int mt = 0; mt < 2; ++mt) {
        v16h af[KS], ag[KS];
        #pragma unroll
        for (int k = 0; k < KS; ++k) {
          af[k] = *(const v16h*)(wp + ((((layer*2 + 0)*2 + mt)*KS + k)*32 + lane)*16);
          ag[k] = *(const v16h*)(wp + ((((layer*2 + 1)*2 + mt)*KS + k)*32 + lane)*16);
        }
        float bfv[8], bgv[8];
        #pragma unroll
        for (int r = 0; r < 8; ++r) {
          int m = mt*16 + hi*8 + r;                 // C-layout row for this VGPR
          bfv[r] = b_f[layer*RES_CH + m];
          bgv[r] = b_g[layer*RES_CH + m];
        }

        for (int nt = 0; nt < NT_PER_WAVE; ++nt) {
          int tl = wave*(NT_PER_WAVE*16) + nt*16 + ln;
          v8f accf, accg;
          #pragma unroll
          for (int r = 0; r < 8; ++r) { accf[r] = bfv[r]; accg[r] = bgv[r]; }
          #pragma unroll
          for (int k = 0; k < KS; ++k) {
            int ts = tl - (KS-1-k)*d;               // causal taps t-2d, t-d, t
            v16h bv = {};
            if (ts >= 0)
              bv = *(const v16h*)(resid + ts*RES_CH + hi*16);
            accf = wmma_f16(af[k], bv, accf);
            accg = wmma_f16(ag[k], bv, accg);
          }
          v8h gv;
          #pragma unroll
          for (int r = 0; r < 8; ++r)
            gv[r] = (_Float16)(fast_tanh(accf[r]) * fast_sigmoid(accg[r]));
          *(v8h*)(gated + tl*RES_CH + mt*16 + hi*8) = gv;   // 8 contiguous ch
        }
      }
    }
    __syncthreads();

    // --- Phase C1: skip_acc += W_s * gated (WMMA; bias folded at the end) ---
    {
      v16h aws[4];
      #pragma unroll
      for (int mt = 0; mt < 4; ++mt)
        aws[mt] = *(const v16h*)(wp + OFF_WS + ((layer*4 + mt)*32 + lane)*16);

      #pragma unroll
      for (int nt = 0; nt < SKIP_NT; ++nt) {
        int tl = HALO + wave*(SKIP_NT*16) + nt*16 + ln;
        v16h bv = *(const v16h*)(gated + tl*RES_CH + hi*16);
        #pragma unroll
        for (int mt = 0; mt < 4; ++mt)
          skipacc[nt][mt] = wmma_f16(aws[mt], bv, skipacc[nt][mt]);
      }
    }

    // --- Phase C2: out += W_o . gated + b_o  (packed-f16 dot per t) ---------
    {
      v8h wh[4];
      #pragma unroll
      for (int q = 0; q < 4; ++q)
        #pragma unroll
        for (int r = 0; r < 8; ++r)
          wh[q][r] = (_Float16)W_o[layer*RES_CH + q*8 + r];
      float bo = b_o[layer];

      float newout[EXT/NTHREADS];
      #pragma unroll
      for (int i2 = 0; i2 < EXT/NTHREADS; ++i2) {
        int t = tid + i2*NTHREADS;
        const v8h* gq = (const v8h*)(gated + t*RES_CH);
        v8h p = gq[0] * wh[0];                       // v_pk_mul/fma_f16
        p += gq[1] * wh[1];
        p += gq[2] * wh[2];
        p += gq[3] * wh[3];
        newout[i2] = outb[t] + bo + hsum8h(p);
      }
      __syncthreads();
      #pragma unroll
      for (int i2 = 0; i2 < EXT/NTHREADS; ++i2)
        outb[tid + i2*NTHREADS] = newout[i2];
    }
    __syncthreads();
  }

  // --- Final head: add folded skip bias, dump to LDS (f16, [t][64]) ---------
  _Float16* skipS = resid;                      // reuse: 512*64 f16 = 64 KiB
  {
    const float* bsum = (const float*)(wp + N_PACK);  // Σ_layers b_s
    float bs4[4][8];
    #pragma unroll
    for (int mt = 0; mt < 4; ++mt)
      #pragma unroll
      for (int r = 0; r < 8; ++r)
        bs4[mt][r] = bsum[mt*16 + hi*8 + r];

    #pragma unroll
    for (int nt = 0; nt < SKIP_NT; ++nt) {
      int tl = wave*(SKIP_NT*16) + nt*16 + ln;  // local tile time 0..511
      #pragma unroll
      for (int mt = 0; mt < 4; ++mt) {
        v8h hv;
        #pragma unroll
        for (int r = 0; r < 8; ++r)
          hv[r] = (_Float16)(skipacc[nt][mt][r] + bs4[mt][r]);
        *(v8h*)(skipS + tl*SKIP_CH + mt*16 + hi*8) = hv;
      }
    }
  }
  __syncthreads();

  // --- skipR = relu(W_sk1 (64x64) * skipS + b_sk1)  (WMMA, K=64 in 2 chunks)
  _Float16* skipR = gated;                      // reuse
  {
    v16h a1[4][2];
    #pragma unroll
    for (int mt = 0; mt < 4; ++mt)
      #pragma unroll
      for (int kc = 0; kc < 2; ++kc)
        a1[mt][kc] = *(const v16h*)(wp + OFF_SK1 + ((mt*2 + kc)*32 + lane)*16);
    float b1v[4][8];
    #pragma unroll
    for (int mt = 0; mt < 4; ++mt)
      #pragma unroll
      for (int r = 0; r < 8; ++r)
        b1v[mt][r] = b_sk1[mt*16 + hi*8 + r];

    #pragma unroll
    for (int nt = 0; nt < SKIP_NT; ++nt) {
      int tl = wave*(SKIP_NT*16) + nt*16 + ln;
      v16h bv0 = *(const v16h*)(skipS + tl*SKIP_CH +      hi*16);
      v16h bv1 = *(const v16h*)(skipS + tl*SKIP_CH + 32 + hi*16);
      #pragma unroll
      for (int mt = 0; mt < 4; ++mt) {
        v8f acc;
        #pragma unroll
        for (int r = 0; r < 8; ++r) acc[r] = b1v[mt][r];
        acc = wmma_f16(a1[mt][0], bv0, acc);
        acc = wmma_f16(a1[mt][1], bv1, acc);
        v8h hv;
        #pragma unroll
        for (int r = 0; r < 8; ++r) hv[r] = (_Float16)fmaxf(acc[r], 0.f);
        *(v8h*)(skipR + tl*SKIP_CH + mt*16 + hi*8) = hv;
      }
    }
  }
  __syncthreads();

  // --- W_sk2 (2x64) head: means + stds (packed-f16 dots) --------------------
  {
    v8h w2m[SKIP_CH/8], w2s[SKIP_CH/8];
    #pragma unroll
    for (int q = 0; q < SKIP_CH/8; ++q)
      #pragma unroll
      for (int r = 0; r < 8; ++r) {
        w2m[q][r] = (_Float16)W_sk2[q*8 + r];
        w2s[q][r] = (_Float16)W_sk2[SKIP_CH + q*8 + r];
      }
    for (int i2 = 0; i2 < TILE/NTHREADS; ++i2) {
      int t = tid + i2*NTHREADS;
      const v8h* cq = (const v8h*)(skipR + t*SKIP_CH);
      v8h pm = cq[0] * w2m[0];
      v8h ps = cq[0] * w2s[0];
      #pragma unroll
      for (int q = 1; q < SKIP_CH/8; ++q) {
        v8h cv = cq[q];
        pm += cv * w2m[q];
        ps += cv * w2s[q];
      }
      float m  = b_sk2[0] + hsum8h(pm);
      float sv = b_sk2[1] + hsum8h(ps);
      long g = (long)bidx*WN_T + t0 + t;
      outp[g] = m;                                   // means
      outp[(long)WN_B*WN_T + g] = __expf(0.5f * sv); // stds
    }
  }
}

// ---------------------------------------------------------------------------
extern "C" void kernel_launch(void* const* d_in, const int* in_sizes, int n_in,
                              void* d_out, int out_size, void* d_ws, size_t ws_size,
                              hipStream_t stream) {
  (void)in_sizes; (void)n_in; (void)out_size; (void)ws_size;
  const float* x     = (const float*)d_in[0];
  const float* W_in  = (const float*)d_in[1];
  const float* b_in  = (const float*)d_in[2];
  const float* W_f   = (const float*)d_in[3];
  const float* b_f   = (const float*)d_in[4];
  const float* W_g   = (const float*)d_in[5];
  const float* b_g   = (const float*)d_in[6];
  const float* W_s   = (const float*)d_in[7];
  const float* b_s   = (const float*)d_in[8];
  const float* W_o   = (const float*)d_in[9];
  const float* b_o   = (const float*)d_in[10];
  const float* W_sk1 = (const float*)d_in[11];
  const float* b_sk1 = (const float*)d_in[12];
  const float* W_sk2 = (const float*)d_in[13];
  const float* b_sk2 = (const float*)d_in[14];
  _Float16* wpack = (_Float16*)d_ws;             // ~265 KiB used
  float* outp = (float*)d_out;

  wn_pack<<<(N_TOTAL + 255)/256, 256, 0, stream>>>(W_f, W_g, W_s, W_sk1, b_s, wpack);

  dim3 grid(WN_B * (WN_T / TILE));               // 1024 workgroups
  wn_fused<<<grid, NTHREADS, SMEM_BYTES, stream>>>(
      x, W_in, b_in, b_f, b_g, W_o, b_o, b_sk1, W_sk2, b_sk2, wpack, outp);
}
